// BaseAttention_51342039056693
// MI455X (gfx1250) — matrix operand order repaired and checked by reading, not compiled, not run on hardware
//
#include <hip/hip_runtime.h>
#include <hip/hip_bf16.h>

// ---------------------------------------------------------------------------
// Linear-attention transformer block for MI455X (gfx1250, wave32, WMMA).
// B=4, L=4096, DM=1024, H=16, D=64.
//
// Pipeline:
//   1) f32->f16 convert of activations + weights
//   2) 3x WMMA GEMM (16384x1024x1024): Q,K with fused elu()+1, V plain
//   3) chunked causal prefix scan (chunk=64): sums -> chunk prefix -> replay
//   4) WMMA GEMM for output projection, fused bias + residual(query)
//   5) LayerNorm -> d_out
//
// GEMM uses double-buffered LDS tiles: one barrier per K-iteration, global
// staging of tile kt+1 overlaps the 8 v_wmma of tile kt, and tile kt+2 is
// prefetched into GL2 via global_prefetch_b8.
// ---------------------------------------------------------------------------

typedef __attribute__((ext_vector_type(16))) _Float16 v16h;
typedef __attribute__((ext_vector_type(8)))  float    v8f;

#define Bb   4
#define Ll   4096
#define DMm  1024
#define Hh   16
#define Dd   64
#define Mdim (Bb * Ll)      // 16384
#define CT   64             // scan chunk length
#define NC   (Ll / CT)      // 64 chunks

// GEMM tiling
#define BM 128
#define BN 128
#define BK 32
#define LDA_DW 20   // A LDS row stride in dwords (32 halves data + pad -> 40 halves; 16B aligned, conflict-free)
#define LDB_DW 66   // B LDS row stride in dwords (128 halves data + pad -> 132 halves; 8B aligned, conflict-free b64)
#define LDSA_SZ (BM * LDA_DW)   // 2560 dwords per buffer
#define LDSB_SZ (BK * LDB_DW)   // 2112 dwords per buffer

union Frag {
    v16h h;
    uint4 u4[2];
    uint2 u2[4];
    unsigned int u[8];
};

// ---------------------------------------------------------------------------
// f32 -> f16 conversion (grid-stride, coalesced; memory bound)
// ---------------------------------------------------------------------------
__global__ void cvt_f32_f16(const float* __restrict__ in, _Float16* __restrict__ out, int n) {
    int i = blockIdx.x * blockDim.x + threadIdx.x;
    int stride = gridDim.x * blockDim.x;
    for (; i < n; i += stride) out[i] = (_Float16)in[i];
}

// ---------------------------------------------------------------------------
// Stage one BMxBK A-tile and BKxBN B-tile from global (f16) into LDS buffers.
// 256 threads: A = 128 rows x 2 segs x 32B; B = 32 rows x 8 segs x 32B.
// ---------------------------------------------------------------------------
__device__ __forceinline__ void stage_tile(
    const _Float16* __restrict__ A, const _Float16* __restrict__ W,
    int K, int N, int bm, int bn, int kt,
    int arow, int aseg, int brow, int bseg,
    unsigned int* __restrict__ dA, unsigned int* __restrict__ dB)
{
    const _Float16* gA = A + (size_t)(bm + arow) * K + kt * BK + aseg * 16;
    uint4 a0 = *(const uint4*)(gA);
    uint4 a1 = *(const uint4*)(gA + 8);
    const _Float16* gB = W + (size_t)(kt * BK + brow) * N + bn + bseg * 16;
    uint4 b0 = *(const uint4*)(gB);
    uint4 b1 = *(const uint4*)(gB + 8);

    unsigned int* da = dA + arow * LDA_DW + aseg * 8;
    *(uint4*)(da)     = a0;
    *(uint4*)(da + 4) = a1;
    unsigned int* db = dB + brow * LDB_DW + bseg * 8;   // 8B-aligned rows (stride 66 dw)
    *(uint2*)(db + 0) = make_uint2(b0.x, b0.y);
    *(uint2*)(db + 2) = make_uint2(b0.z, b0.w);
    *(uint2*)(db + 4) = make_uint2(b1.x, b1.y);
    *(uint2*)(db + 6) = make_uint2(b1.z, b1.w);
}

// ---------------------------------------------------------------------------
// WMMA GEMM: C(M,N) = A(M,K) * W(K,N) [+bias] [+elu1 | +residual]
//   mode 0: C = A*W + bias                    (V projection)
//   mode 1: C = elu(A*W + bias) + 1           (Q, K projections)
//   mode 2: C = A*W + bias + resid            (output projection + residual)
// A, W are f16; C is f32. 256 threads = 8 waves; wave grid 4(M) x 2(N),
// each wave owns a 32x64 subtile = 2x4 v_wmma_f32_16x16x32_f16 accumulators.
// ---------------------------------------------------------------------------
__global__ __launch_bounds__(256) void gemm_wmma_f16(
    const _Float16* __restrict__ A, const _Float16* __restrict__ W,
    const float* __restrict__ bias, const float* __restrict__ resid,
    float* __restrict__ C, int M, int N, int K, int mode)
{
    __shared__ __align__(16) unsigned int ldsA[2 * LDSA_SZ]; // 20480 B
    __shared__ __align__(16) unsigned int ldsB[2 * LDSB_SZ]; // 16896 B

    const int tid    = threadIdx.x;
    const int lane   = tid & 31;
    const int wave   = tid >> 5;
    const int wave_m = wave >> 1;   // 0..3
    const int wave_n = wave & 1;    // 0..1
    const int g      = lane >> 4;   // half-wave group (K-range select for A, M-half for C)
    const int mrow   = lane & 15;

    const int bm = blockIdx.y * BM;
    const int bn = blockIdx.x * BN;

    v8f acc[2][4];
    #pragma unroll
    for (int i = 0; i < 2; ++i)
        #pragma unroll
        for (int j = 0; j < 4; ++j)
            acc[i][j] = (v8f)0.0f;

    // global staging roles
    const int arow = tid >> 1, aseg = tid & 1;   // A: 128 rows x 2 x 32B
    const int brow = tid >> 3, bseg = tid & 7;   // B: 32 rows x 8 x 32B

    const int kTiles = K / BK;

    // prologue: stage tile 0 into buffer 0
    stage_tile(A, W, K, N, bm, bn, 0, arow, aseg, brow, bseg, ldsA, ldsB);
    __syncthreads();

    for (int kt = 0; kt < kTiles; ++kt) {
        const int cur = kt & 1;
        const unsigned int* curA = ldsA + cur * LDSA_SZ;
        const unsigned int* curB = ldsB + cur * LDSB_SZ;

        // overlap: stage tile kt+1 into the other buffer while computing kt.
        // (end-of-iteration barrier made buffer cur^1 free for writing.)
        if (kt + 1 < kTiles)
            stage_tile(A, W, K, N, bm, bn, kt + 1, arow, aseg, brow, bseg,
                       ldsA + (cur ^ 1) * LDSA_SZ, ldsB + (cur ^ 1) * LDSB_SZ);

        // L2 prefetch of the tile after next (global_prefetch_b8 path)
        if (kt + 2 < kTiles) {
            __builtin_prefetch(A + (size_t)(bm + arow) * K + (kt + 2) * BK + aseg * 16, 0, 3);
            __builtin_prefetch(W + (size_t)((kt + 2) * BK + brow) * N + bn + bseg * 16, 0, 3);
        }

        // --- fragments per ISA VGPR layout ---
        Frag afr[2], bfr[4];
        #pragma unroll
        for (int tm = 0; tm < 2; ++tm) {
            const int r = wave_m * 32 + tm * 16 + mrow;
            const unsigned int* pa = &curA[r * LDA_DW + g * 4];
            afr[tm].u4[0] = *(const uint4*)(pa);      // k = g*8 + 0..7
            afr[tm].u4[1] = *(const uint4*)(pa + 8);  // k = 16 + g*8 + 0..7
        }
        #pragma unroll
        for (int tn = 0; tn < 4; ++tn) {
            // lane%16 == N column; elements k = g*8 + 0..7, then 16 + g*8 + 0..7
            const _Float16* pb = (const _Float16*)curB + wave_n * 64 + tn * 16 + mrow;
            #pragma unroll
            for (int s = 0; s < 8; ++s) {
                bfr[tn].h[s]     = pb[(g * 8 + s) * (2 * LDB_DW)];
                bfr[tn].h[8 + s] = pb[(16 + g * 8 + s) * (2 * LDB_DW)];
            }
        }

        #pragma unroll
        for (int tm = 0; tm < 2; ++tm)
            #pragma unroll
            for (int tn = 0; tn < 4; ++tn)
                acc[tm][tn] = __builtin_amdgcn_wmma_f32_16x16x32_f16(
                    false, afr[tm].h, false, bfr[tn].h,
                    (short)0, acc[tm][tn], false, false);

        // single barrier per iteration: publishes buffer cur^1 stores and
        // retires buffer cur reads before it is overwritten at kt+1.
        __syncthreads();
    }

    // --- writeback (C layout: lanes 0-15 -> M=vgpr,N=lane; 16-31 -> M=8+vgpr) ---
    #pragma unroll
    for (int tm = 0; tm < 2; ++tm) {
        #pragma unroll
        for (int tn = 0; tn < 4; ++tn) {
            const int gr = bm + wave_m * 32 + tm * 16 + g * 8;
            const int gc = bn + wave_n * 64 + tn * 16 + mrow;
            const float bv = bias[gc];
            #pragma unroll
            for (int r = 0; r < 8; ++r) {
                const size_t idx = (size_t)(gr + r) * N + gc;
                float val = acc[tm][tn][r] + bv;
                if (mode == 1) val = (val > 0.0f) ? (val + 1.0f) : __expf(val); // elu(x)+1
                if (mode == 2) val += resid[idx];
                C[idx] = val;
            }
        }
    }
}

// ---------------------------------------------------------------------------
// Scan phase 1: per-(b,h,chunk,d) sums of k and k*v over CT=64 timesteps.
// grid = B*H*NC blocks x 64 threads (thread = d). No reductions needed.
// ---------------------------------------------------------------------------
__global__ __launch_bounds__(64) void scan_chunk_sums(
    const float* __restrict__ Kf, const float* __restrict__ Vf,
    float* __restrict__ pK, float* __restrict__ pKV)
{
    const int blk = blockIdx.x;
    const int c  = blk & (NC - 1);
    const int bh = blk >> 6;
    const int b  = bh >> 4, h = bh & (Hh - 1);
    const int d  = threadIdx.x;

    size_t base = ((size_t)(b * Ll + c * CT) * DMm) + h * Dd + d;
    float sk = 0.0f, skv = 0.0f;
    for (int t = 0; t < CT; ++t) {
        float k = Kf[base], v = Vf[base];
        sk += k; skv += k * v;
        base += DMm;
    }
    pK [(size_t)blk * Dd + d] = sk;
    pKV[(size_t)blk * Dd + d] = skv;
}

// ---------------------------------------------------------------------------
// Scan phase 2: in-place exclusive prefix over the NC=64 chunks.
// grid = B*H blocks x 64 threads (thread = d); 64 serial steps, trivial work.
// ---------------------------------------------------------------------------
__global__ __launch_bounds__(64) void scan_chunk_prefix(
    float* __restrict__ pK, float* __restrict__ pKV)
{
    const int bh = blockIdx.x;
    const int d  = threadIdx.x;
    float rk = 0.0f, rkv = 0.0f;
    for (int c = 0; c < NC; ++c) {
        size_t idx = ((size_t)bh * NC + c) * Dd + d;
        float tk = pK[idx], tkv = pKV[idx];
        pK[idx] = rk;  pKV[idx] = rkv;
        rk += tk;      rkv += tkv;
    }
}

// ---------------------------------------------------------------------------
// Scan phase 3: replay chunk with exclusive-prefix carry; emit f16 attn.
// One wave32 per chunk; lane owns channels d and d+32 so the per-timestep
// denominator reduction is a pure intra-wave shfl_xor tree (no barriers).
// ---------------------------------------------------------------------------
__global__ __launch_bounds__(32) void scan_replay(
    const float* __restrict__ Qf, const float* __restrict__ Kf,
    const float* __restrict__ Vf,
    const float* __restrict__ pK, const float* __restrict__ pKV,
    _Float16* __restrict__ attn)
{
    const int blk = blockIdx.x;
    const int c  = blk & (NC - 1);
    const int bh = blk >> 6;
    const int b  = bh >> 4, h = bh & (Hh - 1);
    const int l  = threadIdx.x;           // lane 0..31
    const int d0 = l, d1 = l + 32;

    float Kc0 = pK [(size_t)blk * Dd + d0];
    float Kc1 = pK [(size_t)blk * Dd + d1];
    float KV0 = pKV[(size_t)blk * Dd + d0];
    float KV1 = pKV[(size_t)blk * Dd + d1];

    size_t base = ((size_t)(b * Ll + c * CT) * DMm) + h * Dd;
    for (int t = 0; t < CT; ++t) {
        const size_t i0 = base + d0;
        const float q0 = Qf[i0], q1 = Qf[i0 + 32];
        float p = q0 * Kc0 + q1 * Kc1;
        #pragma unroll
        for (int off = 16; off >= 1; off >>= 1)
            p += __shfl_xor(p, off, 32);
        const float inv = 1.0f / (p + 1e-9f);
        attn[i0]      = (_Float16)(q0 * KV0 * inv);
        attn[i0 + 32] = (_Float16)(q1 * KV1 * inv);
        const float k0 = Kf[i0], k1 = Kf[i0 + 32];
        const float v0 = Vf[i0], v1 = Vf[i0 + 32];
        Kc0 += k0; Kc1 += k1;
        KV0 += k0 * v0; KV1 += k1 * v1;
        base += DMm;
    }
}

// ---------------------------------------------------------------------------
// LayerNorm: one block per row of 1024; block = 256 threads, 4 elems/thread.
// ---------------------------------------------------------------------------
__global__ __launch_bounds__(256) void layernorm_row(
    const float* __restrict__ x, const float* __restrict__ gamma,
    const float* __restrict__ beta, float* __restrict__ out)
{
    const int row = blockIdx.x;
    const int t   = threadIdx.x;
    const int lane = t & 31, wave = t >> 5;
    const float* xr = x + (size_t)row * DMm;

    float v[4], s = 0.0f, ss = 0.0f;
    #pragma unroll
    for (int i = 0; i < 4; ++i) {
        v[i] = xr[t + i * 256];
        s += v[i]; ss += v[i] * v[i];
    }
    #pragma unroll
    for (int off = 16; off >= 1; off >>= 1) {
        s  += __shfl_xor(s,  off, 32);
        ss += __shfl_xor(ss, off, 32);
    }
    __shared__ float red[2][8];
    if (lane == 0) { red[0][wave] = s; red[1][wave] = ss; }
    __syncthreads();
    float S = 0.0f, SS = 0.0f;
    #pragma unroll
    for (int w = 0; w < 8; ++w) { S += red[0][w]; SS += red[1][w]; }

    const float mean = S * (1.0f / DMm);
    const float var  = SS * (1.0f / DMm) - mean * mean;
    const float rstd = rsqrtf(var + 1e-6f);
    #pragma unroll
    for (int i = 0; i < 4; ++i) {
        const int col = t + i * 256;
        out[(size_t)row * DMm + col] = (v[i] - mean) * rstd * gamma[col] + beta[col];
    }
}

// ---------------------------------------------------------------------------
extern "C" void kernel_launch(void* const* d_in, const int* in_sizes, int n_in,
                              void* d_out, int out_size, void* d_ws, size_t ws_size,
                              hipStream_t stream)
{
    const float* query  = (const float*)d_in[0];
    const float* key_in = (const float*)d_in[1];
    const float* value  = (const float*)d_in[2];
    const float* Wq = (const float*)d_in[3];  const float* bq = (const float*)d_in[4];
    const float* Wk = (const float*)d_in[5];  const float* bk = (const float*)d_in[6];
    const float* Wv = (const float*)d_in[7];  const float* bv = (const float*)d_in[8];
    const float* Wo = (const float*)d_in[9];  const float* bo = (const float*)d_in[10];
    const float* gamma = (const float*)d_in[11];
    const float* beta  = (const float*)d_in[12];
    float* out = (float*)d_out;

    const size_t BLDM = (size_t)Mdim * DMm;   // 16,777,216 elems
    const size_t WSZ  = (size_t)DMm * DMm;    //  1,048,576 elems

    // workspace layout (f16 activation region is reused after the scan):
    _Float16* qh  = (_Float16*)d_ws;          // 32MB   (reused as attn_h)
    _Float16* kh  = qh + BLDM;                // 32MB \  (reused as xbuf, 64MB)
    _Float16* vh  = kh + BLDM;                // 32MB /
    _Float16* wqh = vh + BLDM;                // 4 x 2MB f16 weights
    _Float16* wkh = wqh + WSZ;
    _Float16* wvh = wkh + WSZ;
    _Float16* woh = wvh + WSZ;
    float* Qf = (float*)(woh + WSZ);          // 3 x 64MB f32 Q/K/V
    float* Kf = Qf + BLDM;
    float* Vf = Kf + BLDM;
    float* pK  = Vf + BLDM;                   // chunk partials: 2 x 1MB
    float* pKV = pK + (size_t)Bb * Hh * NC * Dd;
    _Float16* attn_h = qh;                    // alias: q-f16 dead after GEMMs
    float*    xbuf   = (float*)kh;            // alias: spans kh+vh (64MB)

    // 1) precision conversion
    cvt_f32_f16<<<4096, 256, 0, stream>>>(query,  qh,  (int)BLDM);
    cvt_f32_f16<<<4096, 256, 0, stream>>>(key_in, kh,  (int)BLDM);
    cvt_f32_f16<<<4096, 256, 0, stream>>>(value,  vh,  (int)BLDM);
    cvt_f32_f16<<<1024, 256, 0, stream>>>(Wq, wqh, (int)WSZ);
    cvt_f32_f16<<<1024, 256, 0, stream>>>(Wk, wkh, (int)WSZ);
    cvt_f32_f16<<<1024, 256, 0, stream>>>(Wv, wvh, (int)WSZ);
    cvt_f32_f16<<<1024, 256, 0, stream>>>(Wo, woh, (int)WSZ);

    // 2) Q/K/V projections (Q,K fused elu+1)
    dim3 gg(DMm / BN, Mdim / BM);             // (8, 128)
    gemm_wmma_f16<<<gg, 256, 0, stream>>>(qh, wqh, bq, nullptr, Qf, Mdim, DMm, DMm, 1);
    gemm_wmma_f16<<<gg, 256, 0, stream>>>(kh, wkh, bk, nullptr, Kf, Mdim, DMm, DMm, 1);
    gemm_wmma_f16<<<gg, 256, 0, stream>>>(vh, wvh, bv, nullptr, Vf, Mdim, DMm, DMm, 0);

    // 3) chunked causal linear-attention scan
    scan_chunk_sums  <<<Bb * Hh * NC, 64, 0, stream>>>(Kf, Vf, pK, pKV);
    scan_chunk_prefix<<<Bb * Hh,      64, 0, stream>>>(pK, pKV);
    scan_replay      <<<Bb * Hh * NC, 32, 0, stream>>>(Qf, Kf, Vf, pK, pKV, attn_h);

    // 4) output projection + bias + residual(query)
    gemm_wmma_f16<<<gg, 256, 0, stream>>>(attn_h, woh, bo, query, xbuf, Mdim, DMm, DMm, 2);

    // 5) LayerNorm
    layernorm_row<<<Mdim, 256, 0, stream>>>(xbuf, gamma, beta, out);
}
